// Linear_87729001988165
// MI455X (gfx1250) — matrix operand order repaired and checked by reading, not compiled, not run on hardware
//
#include <hip/hip_runtime.h>

// ---- CDNA5 (gfx1250) wave32 WMMA / TDM types ----
typedef __bf16        v16bf __attribute__((ext_vector_type(16)));
typedef float         v8f   __attribute__((ext_vector_type(8)));
typedef unsigned int  u32x4 __attribute__((ext_vector_type(4)));
typedef unsigned int  u32x2 __attribute__((ext_vector_type(2)));
typedef int           i32x4 __attribute__((ext_vector_type(4)));
typedef int           i32x8 __attribute__((ext_vector_type(8)));

union ABFrag { u32x4 q[2]; v16bf v; };

// round-to-nearest-even f32 -> bf16 (bit pattern), packed pair
__device__ __forceinline__ unsigned int pack_bf16x2(float lo, float hi) {
    unsigned int ulo = __float_as_uint(lo);
    unsigned int uhi = __float_as_uint(hi);
    ulo += 0x7FFFu + ((ulo >> 16) & 1u);
    uhi += 0x7FFFu + ((uhi >> 16) & 1u);
    return (ulo >> 16) | (uhi & 0xFFFF0000u);
}

// =====================================================================
// Phase 1a: x (f32) -> bf16 workspace
// =====================================================================
__global__ __launch_bounds__(256)
void cvt_x_bf16(const float* __restrict__ x, unsigned short* __restrict__ xb,
                size_t n4 /* float4 count */) {
    size_t i = (size_t)blockIdx.x * 256 + threadIdx.x;
    if (i >= n4) return;
    float4 v = ((const float4*)x)[i];
    u32x2 p;
    p.x = pack_bf16x2(v.x, v.y);
    p.y = pack_bf16x2(v.z, v.w);
    ((u32x2*)xb)[i] = p;
}

// =====================================================================
// Phase 1b: weight (f32) * block-scale -> bf16 workspace
// grid = (IN/1024, OUT); each thread handles one float4 of one row
// =====================================================================
__global__ __launch_bounds__(256)
void cvt_w_bf16(const float* __restrict__ w, const float* __restrict__ scale,
                unsigned short* __restrict__ wb, int IN, int scale_cols) {
    const int o  = blockIdx.y;
    const int c4 = blockIdx.x * 256 + threadIdx.x;       // float4 col
    const float s = scale[(size_t)(o >> 6) * scale_cols + (c4 >> 4)]; // (c4*4)>>6
    const size_t i = (size_t)o * (IN >> 2) + c4;
    float4 v = ((const float4*)w)[i];
    u32x2 p;
    p.x = pack_bf16x2(v.x * s, v.y * s);
    p.y = pack_bf16x2(v.z * s, v.w * s);
    ((u32x2*)wb)[i] = p;
}

// =====================================================================
// TDM 2D tile load: 32 (K, contiguous bf16) x 128 rows, row stride = stride_elems.
// LDS destination rows padded 64B data + 16B pad -> 80B stride (conflict-free
// b128 fragment reads: start banks 20*m mod 64 are distinct multiples of 4).
// Descriptor per CDNA5 ISA section 8 (D#).
// =====================================================================
__device__ __forceinline__ void tdm_load_tile(unsigned lds_off, const void* gptr,
                                              unsigned stride_elems) {
    unsigned long long ga = (unsigned long long)(size_t)gptr;
    u32x4 g0;
    g0.x = 1u;                                              // count=1, user mode
    g0.y = lds_off;                                         // lds_addr (bytes)
    g0.z = (unsigned)ga;                                    // global_addr[31:0]
    g0.w = (unsigned)((ga >> 32) & 0x01FFFFFFull)           // global_addr[56:32]
         | (2u << 30);                                      // type=2 ("image")
    i32x8 g1;
    g1[0] = (1 << 16)      // data_size = 1 -> 2 bytes
          | (1 << 20)      // pad_enable
          | (3 << 22)      // pad_interval: 16 DWORDs (=64B row) before padding
          | (3 << 25);     // pad_amount:   4 DWORDs (=16B)
    g1[1] = 0;                         // abar_addr=0 | tensor_dim0[15:0] (2^20 -> 0)
    g1[2] = 16;                        // tensor_dim0[31:16]=16 | tensor_dim1[15:0]=0
    g1[3] = 16 | (32 << 16);           // tensor_dim1[31:16]=16 | tile_dim0=32
    g1[4] = 128;                       // tile_dim1=128 | tile_dim2=0
    g1[5] = (int)stride_elems;         // tensor_dim0_stride[31:0]
    g1[6] = 0;                         // stride[47:32]=0 | dim1_stride lo = 0
    g1[7] = 0;
    i32x4 z4 = {0, 0, 0, 0};
#if __clang_major__ >= 23
    i32x8 z8 = {0, 0, 0, 0, 0, 0, 0, 0};
    __builtin_amdgcn_tensor_load_to_lds(g0, g1, z4, z4, z8, 0);
#else
    __builtin_amdgcn_tensor_load_to_lds(g0, g1, z4, z4, 0);
#endif
}

// =====================================================================
// Phase 2: bf16 GEMM, TDM double-buffered global->LDS, WMMA compute.
// 128x128 tile / WG, K-slab 32; 8 waves in 4(M) x 2(N) grid, 2x4 WMMA each.
// Dynamic LDS: 2 buffers x (x-tile + w-tile) x 128 rows x 80B = 40960 B.
// =====================================================================
#define ROWB   80
#define TILE_B (128 * ROWB)   // 10240 bytes per tile

__global__ __launch_bounds__(256, 2)
void gemm_tdm_wmma_bf16(const unsigned short* __restrict__ xb,
                        const unsigned short* __restrict__ wb,
                        float* __restrict__ out, int T, int IN, int OUT)
{
    extern __shared__ char smem[];   // [buf0: x,w][buf1: x,w]

    const int tid  = threadIdx.x;
    const int lane = tid & 31;
    const int wv   = tid >> 5;
    const int wm   = wv & 3;
    const int wn   = wv >> 2;
    const int lm   = lane & 15;
    const int lh   = lane >> 4;

    const int t0 = blockIdx.y * 128;
    const int o0 = blockIdx.x * 128;

    const unsigned short* xrow = xb + (size_t)t0 * IN;
    const unsigned short* wrow = wb + (size_t)o0 * IN;

    v8f acc[2][4] = {};
    const int nK = IN >> 5;

    if (wv == 0) {                       // one wave drives the TDM
        tdm_load_tile(0,      xrow, (unsigned)IN);
        tdm_load_tile(TILE_B, wrow, (unsigned)IN);
    }

    for (int kb = 0; kb < nK; ++kb) {
        const int cur = kb & 1;
        if (wv == 0) {
            if (kb + 1 < nK) {
                const unsigned nb = (unsigned)((kb + 1) & 1) * (2 * TILE_B);
                tdm_load_tile(nb,          xrow + (size_t)(kb + 1) * 32, (unsigned)IN);
                tdm_load_tile(nb + TILE_B, wrow + (size_t)(kb + 1) * 32, (unsigned)IN);
                __builtin_amdgcn_s_wait_tensorcnt(2);  // current stage's 2 ops done
            } else {
                __builtin_amdgcn_s_wait_tensorcnt(0);
            }
        }
        __syncthreads();                  // stage data visible to all waves

        const char* bx = smem + (size_t)cur * (2 * TILE_B);
        const char* bw = bx + TILE_B;

        // A fragments: lane L elem h <-> (M=L%16, K=16*(h/8)+8*(L/16)+h%8)
        v16bf a[2];
        #pragma unroll
        for (int mi = 0; mi < 2; ++mi) {
            const int row = wm * 32 + mi * 16 + lm;
            const u32x4* p = (const u32x4*)(bx + row * ROWB);
            ABFrag f;
            f.q[0] = p[lh];
            f.q[1] = p[2 + lh];
            a[mi] = f.v;
        }
        // B fragments: lane L elem h <-> (N=L%16, K=16*(h/8)+8*(L/16)+h%8)
        v16bf b[4];
        #pragma unroll
        for (int ni = 0; ni < 4; ++ni) {
            const int row = wn * 64 + ni * 16 + lm;
            const u32x4* p = (const u32x4*)(bw + row * ROWB);
            ABFrag f;
            f.q[0] = p[lh];
            f.q[1] = p[2 + lh];
            b[ni] = f.v;
        }

        #pragma unroll
        for (int mi = 0; mi < 2; ++mi)
            #pragma unroll
            for (int ni = 0; ni < 4; ++ni)
                acc[mi][ni] = __builtin_amdgcn_wmma_f32_16x16x32_bf16(
                    false, a[mi], false, b[ni],
                    (short)0, acc[mi][ni], false, false);

        __syncthreads();                  // buffer free for next TDM write
    }

    // C/D layout: VGPR r, lanes 0-15 -> M=r, lanes 16-31 -> M=8+r; N=lane%16
    #pragma unroll
    for (int mi = 0; mi < 2; ++mi) {
        #pragma unroll
        for (int ni = 0; ni < 4; ++ni) {
            const int row0 = t0 + wm * 32 + mi * 16 + lh * 8;
            const int col  = o0 + wn * 64 + ni * 16 + lm;
            float* op = &out[(size_t)row0 * OUT + col];
            #pragma unroll
            for (int r = 0; r < 8; ++r)
                op[(size_t)r * OUT] = acc[mi][ni][r];
        }
    }
}

// =====================================================================
// Fallback (round-1 kernel): fused convert-in-loop path, used only if
// the workspace is too small for the bf16 panels.
// =====================================================================
#define LDS_STRIDE 40

__global__ __launch_bounds__(256, 2)
void linear_dq_wmma_fallback(const float* __restrict__ x,
                             const float* __restrict__ w,
                             const float* __restrict__ scale,
                             float* __restrict__ out,
                             int T, int IN, int OUT, int scale_cols)
{
    __shared__ __align__(16) unsigned short lx[128 * LDS_STRIDE];
    __shared__ __align__(16) unsigned short lw[128 * LDS_STRIDE];

    const int tid  = threadIdx.x;
    const int lane = tid & 31;
    const int wv   = tid >> 5;
    const int wm   = wv & 3;
    const int wn   = wv >> 2;
    const int lm   = lane & 15;
    const int lh   = lane >> 4;

    const int t0 = blockIdx.y * 128;
    const int o0 = blockIdx.x * 128;

    const int ldRow  = tid >> 3;
    const int ldCol4 = tid & 7;

    v8f acc[2][4] = {};

    const int nK = IN >> 5;
    for (int kb = 0; kb < nK; ++kb) {
        const int k0 = kb << 5;
        float4 rx[4], rw[4];
        float  rs[4];
        #pragma unroll
        for (int p = 0; p < 4; ++p) {
            const int r = ldRow + p * 32;
            rx[p] = *(const float4*)&x[(size_t)(t0 + r) * IN + k0 + ldCol4 * 4];
            const int o = o0 + r;
            rw[p] = *(const float4*)&w[(size_t)o * IN + k0 + ldCol4 * 4];
            rs[p] = scale[(size_t)(o >> 6) * scale_cols + (k0 >> 6)];
        }
        __syncthreads();
        #pragma unroll
        for (int p = 0; p < 4; ++p) {
            const int r = ldRow + p * 32;
            u32x2 px, pw;
            px.x = pack_bf16x2(rx[p].x, rx[p].y);
            px.y = pack_bf16x2(rx[p].z, rx[p].w);
            const float s = rs[p];
            pw.x = pack_bf16x2(rw[p].x * s, rw[p].y * s);
            pw.y = pack_bf16x2(rw[p].z * s, rw[p].w * s);
            *(u32x2*)&lx[r * LDS_STRIDE + ldCol4 * 4] = px;
            *(u32x2*)&lw[r * LDS_STRIDE + ldCol4 * 4] = pw;
        }
        __syncthreads();

        v16bf a[2];
        #pragma unroll
        for (int mi = 0; mi < 2; ++mi) {
            const int row = wm * 32 + mi * 16 + lm;
            const u32x4* p = (const u32x4*)&lx[row * LDS_STRIDE];
            ABFrag f; f.q[0] = p[lh]; f.q[1] = p[2 + lh];
            a[mi] = f.v;
        }
        v16bf b[4];
        #pragma unroll
        for (int ni = 0; ni < 4; ++ni) {
            const int row = wn * 64 + ni * 16 + lm;
            const u32x4* p = (const u32x4*)&lw[row * LDS_STRIDE];
            ABFrag f; f.q[0] = p[lh]; f.q[1] = p[2 + lh];
            b[ni] = f.v;
        }
        #pragma unroll
        for (int mi = 0; mi < 2; ++mi)
            #pragma unroll
            for (int ni = 0; ni < 4; ++ni)
                acc[mi][ni] = __builtin_amdgcn_wmma_f32_16x16x32_bf16(
                    false, a[mi], false, b[ni],
                    (short)0, acc[mi][ni], false, false);
    }

    #pragma unroll
    for (int mi = 0; mi < 2; ++mi) {
        #pragma unroll
        for (int ni = 0; ni < 4; ++ni) {
            const int row0 = t0 + wm * 32 + mi * 16 + lh * 8;
            const int col  = o0 + wn * 64 + ni * 16 + lm;
            float* op = &out[(size_t)row0 * OUT + col];
            #pragma unroll
            for (int r = 0; r < 8; ++r)
                op[(size_t)r * OUT] = acc[mi][ni][r];
        }
    }
}

// =====================================================================
extern "C" void kernel_launch(void* const* d_in, const int* in_sizes, int n_in,
                              void* d_out, int out_size, void* d_ws, size_t ws_size,
                              hipStream_t stream) {
    const float* x     = (const float*)d_in[0];
    const float* w     = (const float*)d_in[1];
    const float* scale = (const float*)d_in[2];
    float* out = (float*)d_out;

    const int IN  = 4096;
    const int T   = in_sizes[0] / IN;     // 8192
    const int OUT = in_sizes[1] / IN;     // 11008
    const int scale_cols = IN / 64;       // 64

    const size_t need = ((size_t)T + (size_t)OUT) * (size_t)IN * 2; // bf16 panels
    dim3 grid(OUT / 128, T / 128);

    if (ws_size >= need) {
        unsigned short* xb = (unsigned short*)d_ws;
        unsigned short* wb = xb + (size_t)T * IN;

        const size_t n4x = (size_t)T * IN / 4;
        cvt_x_bf16<<<(unsigned)((n4x + 255) / 256), 256, 0, stream>>>(x, xb, n4x);
        cvt_w_bf16<<<dim3(IN / 1024, OUT), 256, 0, stream>>>(w, scale, wb, IN, scale_cols);

        gemm_tdm_wmma_bf16<<<grid, 256, 4 * TILE_B, stream>>>(xb, wb, out, T, IN, OUT);
    } else {
        linear_dq_wmma_fallback<<<grid, 256, 0, stream>>>(x, w, scale, out,
                                                          T, IN, OUT, scale_cols);
    }
}
